// QKVAttention_49598282334655
// MI455X (gfx1250) — matrix operand order repaired and checked by reading, not compiled, not run on hardware
//
#include <hip/hip_runtime.h>

typedef __bf16 v16bf __attribute__((ext_vector_type(16)));
typedef float  v8f   __attribute__((ext_vector_type(8)));

#define WMMA_BF16(A, Bm, C) \
  __builtin_amdgcn_wmma_f32_16x16x32_bf16(false, (A), false, (Bm), (short)0, (C), false, false)

constexpr int   kB     = 8;
constexpr int   kS     = 2048;
constexpr int   kH     = 512;
constexpr float kScale = 0.04419417382415922f;  // 1/sqrt(512)
constexpr int   QT     = 32;                    // queries per workgroup
constexpr int   KT     = 32;                    // keys per tile
constexpr int   NW     = 8;                     // waves per workgroup
constexpr int   HW     = kH / NW;               // 64 H columns per wave

__global__ __launch_bounds__(256)
void flash_attn_bf16_wmma(const float* __restrict__ Q,
                          const float* __restrict__ K,
                          const float* __restrict__ V,
                          const int*   __restrict__ vlen,
                          float* __restrict__ Out) {
  __shared__ float  sP[NW * QT * KT];   // per-wave partial scores (32 KB)
  __shared__ float  sRed[QT * KT];      // reduced scores (4 KB)
  __shared__ __bf16 sProb[QT * KT];     // probabilities bf16 (2 KB)
  __shared__ float  sC[QT];             // per-row online-softmax rescale
  __shared__ float  sL[QT];             // per-row denominator
  __shared__ int    sVl[QT];

  const int tid  = threadIdx.x;
  const int lane = tid & 31;
  const int w    = tid >> 5;
  const int half = lane >> 4;           // 0: lanes 0-15, 1: lanes 16-31
  const int ln16 = lane & 15;

  const int blk = blockIdx.x;
  const int b   = blk / (kS / QT);
  const int q0  = (blk % (kS / QT)) * QT;
  const int hw  = w * HW;

  // ---- per-row valid length; block max -> number of key tiles
  const int vl = vlen[(size_t)b * kS + q0 + lane];
  if (tid < QT) sVl[tid] = vlen[(size_t)b * kS + q0 + tid];
  __syncthreads();
  int maxvl = 1;
  #pragma unroll
  for (int i = 0; i < QT; ++i) maxvl = max(maxvl, sVl[i]);
  const int nkt = (maxvl + KT - 1) / KT;

  // ---- Q tiles, A layout (16x32, contiguous k = 16*half + s), resident all kernel
  v16bf Qa[2][2];
  #pragma unroll
  for (int qs = 0; qs < 2; ++qs) {
    const float* qrow = Q + (size_t)(b * kS + q0 + qs * 16 + ln16) * kH + hw;
    #pragma unroll
    for (int hc = 0; hc < 2; ++hc) {
      #pragma unroll
      for (int p = 0; p < 8; ++p) {
        const int k0 = 16 * half + 2 * p;
        Qa[qs][hc][2 * p]     = (__bf16)qrow[hc * 32 + k0];
        Qa[qs][hc][2 * p + 1] = (__bf16)qrow[hc * 32 + k0 + 1];
      }
    }
  }

  v8f Ozero;
  #pragma unroll
  for (int r = 0; r < 8; ++r) Ozero[r] = 0.0f;
  v8f O[2][4];
  #pragma unroll
  for (int qs = 0; qs < 2; ++qs)
    #pragma unroll
    for (int ht = 0; ht < 4; ++ht) O[qs][ht] = Ozero;

  float mrow = -1e30f, lrow = 0.0f;  // online-softmax state (wave 0 only)

  for (int kt = 0; kt < nkt; ++kt) {
    const int kb = kt * KT;

    // ---- K tiles, B layout (32x16: lane = key column, K-dim = H chunk)
    v16bf Kb[2][2];
    #pragma unroll
    for (int ks = 0; ks < 2; ++ks) {
      const float* krow = K + (size_t)(b * kS + kb + ks * 16 + ln16) * kH + hw;
      #pragma unroll
      for (int hc = 0; hc < 2; ++hc) {
        const float* p16 = krow + hc * 32 + 16 * half;  // 16 contiguous floats
        #pragma unroll
        for (int e = 0; e < 16; ++e) Kb[ks][hc][e] = (__bf16)p16[e];
      }
    }

    // ---- partial scores over this wave's 64-wide H slice
    #pragma unroll
    for (int qs = 0; qs < 2; ++qs) {
      #pragma unroll
      for (int ks = 0; ks < 2; ++ks) {
        v8f acc = Ozero;
        acc = WMMA_BF16(Qa[qs][0], Kb[ks][0], acc);
        acc = WMMA_BF16(Qa[qs][1], Kb[ks][1], acc);
        #pragma unroll
        for (int r = 0; r < 8; ++r)
          sP[w * QT * KT + (qs * 16 + r + 8 * half) * KT + ks * 16 + ln16] = acc[r];
      }
    }

    // ---- prefetch V tiles (B layout) before the barriers to hide latency
    v16bf Vb[4];
    #pragma unroll
    for (int ht = 0; ht < 4; ++ht) {
      const float* vb = V + (size_t)(b * kS + kb + 16 * half) * kH + hw + ht * 16 + ln16;
      #pragma unroll
      for (int p = 0; p < 8; ++p) {
        Vb[ht][2 * p]     = (__bf16)vb[(size_t)(2 * p) * kH];
        Vb[ht][2 * p + 1] = (__bf16)vb[(size_t)(2 * p + 1) * kH];
      }
    }

    __syncthreads();

    // ---- reduce partial scores across the 8 waves (all 256 threads)
    #pragma unroll
    for (int e = 0; e < 4; ++e) {
      const int idx = e * 256 + tid;
      float s = 0.0f;
      #pragma unroll
      for (int ww = 0; ww < NW; ++ww) s += sP[ww * QT * KT + idx];
      sRed[idx] = s;
    }
    __syncthreads();

    // ---- masked online softmax: wave 0, lane = query row
    if (w == 0) {
      const int row = lane;
      float tm = -1e30f;
      #pragma unroll
      for (int j = 0; j < KT; ++j) {
        const float s = sRed[row * KT + j] * kScale;
        tm = ((kb + j) < vl) ? fmaxf(tm, s) : tm;
      }
      const float mnew = fmaxf(mrow, tm);
      const float corr = __expf(mrow - mnew);   // 0 on first tile (mrow=-1e30)
      float psum = 0.0f;
      #pragma unroll
      for (int j = 0; j < KT; ++j) {
        const float s  = sRed[row * KT + j] * kScale;
        const float pj = ((kb + j) < vl) ? __expf(s - mnew) : 0.0f;
        psum += pj;
        sProb[row * KT + j] = (__bf16)pj;
      }
      lrow = corr * lrow + psum;
      mrow = mnew;
      sC[row] = corr;
    }
    __syncthreads();

    // ---- load P tiles (A layout) from LDS
    v16bf Pa[2];
    #pragma unroll
    for (int qs = 0; qs < 2; ++qs) {
      #pragma unroll
      for (int p = 0; p < 8; ++p) {
        const int k0 = 16 * half + 2 * p;
        Pa[qs][2 * p]     = sProb[(qs * 16 + ln16) * KT + k0];
        Pa[qs][2 * p + 1] = sProb[(qs * 16 + ln16) * KT + k0 + 1];
      }
    }

    // ---- rescale running output, accumulate P x V on this wave's H slice
    #pragma unroll
    for (int qs = 0; qs < 2; ++qs) {
      #pragma unroll
      for (int r = 0; r < 8; ++r) {
        const float c = sC[qs * 16 + r + 8 * half];
        #pragma unroll
        for (int ht = 0; ht < 4; ++ht) O[qs][ht][r] *= c;
      }
      #pragma unroll
      for (int ht = 0; ht < 4; ++ht)
        O[qs][ht] = WMMA_BF16(Pa[qs], Vb[ht], O[qs][ht]);
    }
    // no barrier needed here: next iteration's sP writes are fenced by
    // this iteration's B2/B3 barriers relative to all shared readers.
  }

  if (w == 0) sL[lane] = lrow;
  __syncthreads();

  // ---- normalize and store (C/D layout: VGPR r -> row r + 8*half)
  #pragma unroll
  for (int qs = 0; qs < 2; ++qs) {
    #pragma unroll
    for (int r = 0; r < 8; ++r) {
      const int row  = qs * 16 + r + 8 * half;
      const float iv = 1.0f / sL[row];
      float* orow = Out + (size_t)(b * kS + q0 + row) * kH + hw + ln16;
      #pragma unroll
      for (int ht = 0; ht < 4; ++ht) orow[ht * 16] = O[qs][ht][r] * iv;
    }
  }
}

extern "C" void kernel_launch(void* const* d_in, const int* in_sizes, int n_in,
                              void* d_out, int out_size, void* d_ws, size_t ws_size,
                              hipStream_t stream) {
  const float* Q  = (const float*)d_in[0];
  const float* K  = (const float*)d_in[1];
  const float* V  = (const float*)d_in[2];
  const int*   vl = (const int*)d_in[3];
  float* out = (float*)d_out;

  dim3 grid(kB * (kS / QT));   // 8 * 64 = 512 workgroups
  dim3 block(256);             // 8 waves (wave32)
  hipLaunchKernelGGL(flash_attn_bf16_wmma, grid, block, 0, stream,
                     Q, K, V, vl, out);
}
